// DiffusionAdapterLayer_29119878267609
// MI455X (gfx1250) — matrix operand order repaired and checked by reading, not compiled, not run on hardware
//
#include <hip/hip_runtime.h>
#include <hip/hip_bf16.h>

typedef __attribute__((ext_vector_type(16))) _Float16 v16h;
typedef __attribute__((ext_vector_type(8)))  _Float16 v8h;
typedef __attribute__((ext_vector_type(8)))  float    v8f;

// Async-copy builtin wants: (global int4*, shared int4*, imm offset, imm cpol)
typedef int v4i_raw __attribute__((vector_size(16)));
typedef __attribute__((address_space(1))) v4i_raw* gv4i_t;
typedef __attribute__((address_space(3))) v4i_raw* lv4i_t;

#define B_   64
#define C_   1024
#define T_   512
#define CB_  256
#define G_   8
#define CG_  128          /* C_/G_ */
#define NTOT (B_ * T_)    /* 32768 */
#define EPS_ 1e-5f

#define KC_    64                 /* K-chunk staged in LDS per buffer       */
#define LDSTR  (KC_ + 8)          /* padded LDS row stride in halves       */
#define BUFH   (128 * LDSTR)      /* halves per LDS buffer                 */

__device__ __forceinline__ float mishf(float v) {
  float sp = (v > 20.f) ? v : __logf(1.f + __expf(v));
  return v * tanhf(sp);
}

__device__ __forceinline__ void wait_async_all() {
#if __has_builtin(__builtin_amdgcn_s_wait_asynccnt)
  __builtin_amdgcn_s_wait_asynccnt(0);
#else
  asm volatile("s_wait_asynccnt 0x0" ::: "memory");
#endif
}

// ---------------------------------------------------------------------------
// 1) GroupNorm statistics: one block per (b, g); group = 128ch x 512T
// ---------------------------------------------------------------------------
__global__ __launch_bounds__(256) void gn_stats_kernel(
    const float* __restrict__ x, float* __restrict__ mr /* [B_*G_][2] */) {
  const int bg = blockIdx.x;                 // 0..511
  const int tid = threadIdx.x;
  const float* p = x + (size_t)bg * CG_ * T_; // groups contiguous in [B,C,T]
  const float4* p4 = (const float4*)p;
  float s = 0.f, ss = 0.f;
  for (int i = tid; i < (CG_ * T_) / 4; i += 256) {
    float4 v = p4[i];
    s  += v.x + v.y + v.z + v.w;
    ss += v.x * v.x + v.y * v.y + v.z * v.z + v.w * v.w;
  }
  __shared__ float sh[512];
  sh[tid] = s; sh[tid + 256] = ss;
  __syncthreads();
  for (int off = 128; off > 0; off >>= 1) {
    if (tid < off) { sh[tid] += sh[tid + off]; sh[tid + 256] += sh[tid + 256 + off]; }
    __syncthreads();
  }
  if (tid == 0) {
    const float inv = 1.0f / (float)(CG_ * T_);
    float mean = sh[0] * inv;
    float var  = sh[256] * inv - mean * mean;
    mr[2 * bg]     = mean;
    mr[2 * bg + 1] = rsqrtf(var + EPS_);
  }
}

// ---------------------------------------------------------------------------
// 2) Normalize + affine, cast f16, transpose [B,C,T] -> [B*T, C] via LDS tile
// ---------------------------------------------------------------------------
__global__ __launch_bounds__(256) void gn_apply_t_kernel(
    const float* __restrict__ x, const float* __restrict__ gamma,
    const float* __restrict__ beta, const float* __restrict__ mr,
    _Float16* __restrict__ hn /* [NTOT, C_] */) {
  const int t0 = blockIdx.x * 32;
  const int c0 = blockIdx.y * 32;
  const int b  = blockIdx.z;
  __shared__ _Float16 tile[32][33];
  const int tx = threadIdx.x & 31;
  const int ty = threadIdx.x >> 5;           // 0..7
  const int g  = c0 / CG_;
  const float mean = mr[2 * (b * G_ + g)];
  const float rstd = mr[2 * (b * G_ + g) + 1];
#pragma unroll
  for (int i = 0; i < 4; ++i) {
    const int c = c0 + ty + 8 * i;
    const float v  = x[((size_t)b * C_ + c) * T_ + t0 + tx];
    const float nv = (v - mean) * rstd * gamma[c] + beta[c];
    tile[ty + 8 * i][tx] = (_Float16)nv;
  }
  __syncthreads();
#pragma unroll
  for (int i = 0; i < 4; ++i) {
    const int t = t0 + ty + 8 * i;
    hn[((size_t)b * T_ + t) * C_ + c0 + tx] = tile[tx][ty + 8 * i];
  }
}

// ---------------------------------------------------------------------------
// 3) f32 -> f16 weight conversion
// ---------------------------------------------------------------------------
__global__ __launch_bounds__(256) void cvt_f16_kernel(
    const float* __restrict__ src, _Float16* __restrict__ dst, int n) {
  int i = blockIdx.x * 256 + threadIdx.x;
  if (i < n) dst[i] = (_Float16)src[i];
}

// ---------------------------------------------------------------------------
// Async stage of one [128 x KC_] weight chunk into a padded LDS buffer.
// 256 threads x 4 x b128 (16B) = 128 rows x 64 halves. Tracked by ASYNCcnt.
// ---------------------------------------------------------------------------
template <int K>
__device__ __forceinline__ void stage_chunk(const _Float16* __restrict__ A,
                                            _Float16* __restrict__ buf,
                                            int m_base, int kc, int tid) {
#pragma unroll
  for (int i = 0; i < 4; ++i) {
    const int li = tid + 256 * i;       // 0..1023
    const int r  = li >> 3;             // row 0..127
    const int c8 = (li & 7) * 8;        // half-offset 0,8,...,56
    const _Float16* g = A + (size_t)(m_base + r) * K + kc + c8;
    _Float16* l = buf + r * LDSTR + c8;
#if __has_builtin(__builtin_amdgcn_global_load_async_to_lds_b128)
    __builtin_amdgcn_global_load_async_to_lds_b128(
        (gv4i_t)(uintptr_t)(const void*)g,
        (lv4i_t)(uint32_t)(uintptr_t)(void*)l, 0, 0);
#else
    *(v8h*)l = *(const v8h*)g;
#endif
  }
}

// ---------------------------------------------------------------------------
// Shared WMMA mainloop: wave computes 128(M) x 16(N); A staged via double-
// buffered async LDS; B streamed from global (K-contiguous [N,K] layout).
// ---------------------------------------------------------------------------
template <int K>
__device__ __forceinline__ void gemm_core(const _Float16* __restrict__ A,
                                          const _Float16* __restrict__ Bm,
                                          _Float16* __restrict__ smem, // [2*BUFH]
                                          int m_base, int n0, int tid,
                                          v8f acc[8]) {
  const int lane = tid & 31;
  const int hi = lane >> 4;
  const int lo = lane & 15;
  const _Float16* brow = Bm + (size_t)(n0 + lo) * K + hi * 16;
  constexpr int NCHUNK = K / KC_;

  stage_chunk<K>(A, smem, m_base, 0, tid);
  wait_async_all();
  __syncthreads();

  for (int c = 0; c < NCHUNK; ++c) {
    const _Float16* buf = smem + (c & 1) * BUFH;
    if (c + 1 < NCHUNK)
      stage_chunk<K>(A, smem + ((c + 1) & 1) * BUFH, m_base, (c + 1) * KC_, tid);

#pragma unroll
    for (int kk = 0; kk < KC_; kk += 32) {
      const int k0 = c * KC_ + kk;
      v16h bf = *(const v16h*)(brow + k0);
      if (k0 + 32 < K) __builtin_prefetch(brow + k0 + 32, 0, 3);

      // Preload all 8 A fragments from LDS (conflict-free padded layout),
      // then issue the 8-WMMA chain so DS reads overlap matrix issue.
      v16h af[8];
#pragma unroll
      for (int mt = 0; mt < 8; ++mt) {
        const _Float16* ap = buf + (mt * 16 + lo) * LDSTR + kk;
        v8h a_lo = *(const v8h*)(ap + hi * 16);
        v8h a_hi = *(const v8h*)(ap + hi * 16 + 8);
        af[mt] = __builtin_shufflevector(a_lo, a_hi, 0, 1, 2, 3, 4, 5, 6, 7,
                                         8, 9, 10, 11, 12, 13, 14, 15);
      }
#pragma unroll
      for (int mt = 0; mt < 8; ++mt)
        acc[mt] = __builtin_amdgcn_wmma_f32_16x16x32_f16(
            false, af[mt], false, bf, (short)0, acc[mt], false, false);
    }

    if (c + 1 < NCHUNK) { wait_async_all(); __syncthreads(); }
  }
}

// ---------------------------------------------------------------------------
// 4) GEMM down + bias + mish:  D[n, o] = mish(sum_c Bm[n,c] * A[o,c] + bias)
// ---------------------------------------------------------------------------
__global__ __launch_bounds__(256) void gemm_down_mish_kernel(
    const _Float16* __restrict__ A, const _Float16* __restrict__ Bm,
    const float* __restrict__ bias, _Float16* __restrict__ D) {
  __shared__ _Float16 smem[2 * BUFH];
  const int tid = threadIdx.x;
  const int lane = tid & 31;
  const int wave = tid >> 5;
  const int hi = lane >> 4;
  const int lo = lane & 15;
  const int n0 = blockIdx.x * 128 + wave * 16;
  const int m_base = blockIdx.y * 128;

  v8f acc[8];
#pragma unroll
  for (int i = 0; i < 8; ++i) acc[i] = (v8f)0.f;

  gemm_core<C_>(A, Bm, smem, m_base, n0, tid, acc);

  // acc layout: lanes 0-15: M=vgpr, N=lane; lanes 16-31: M=vgpr+8, N=lane-16.
  const int nn = n0 + lo;
#pragma unroll
  for (int mt = 0; mt < 8; ++mt) {
    const int ob = m_base + mt * 16 + hi * 8;
    v8h ov;
#pragma unroll
    for (int j = 0; j < 8; ++j) {
      float v = acc[mt][j] + bias[ob + j];
      ov[j] = (_Float16)mishf(v);
    }
    *(v8h*)(D + (size_t)nn * CB_ + ob) = ov;
  }
}

// ---------------------------------------------------------------------------
// 5) GEMM up + bias + residual:
//    out[b,o,t] = sum_cb D1[n,cb] * A[o,cb] + bias[o] + x[b,o,t],  n=b*T+t
// ---------------------------------------------------------------------------
__global__ __launch_bounds__(256) void gemm_up_res_kernel(
    const _Float16* __restrict__ A, const _Float16* __restrict__ Bm,
    const float* __restrict__ bias, const float* __restrict__ xres,
    float* __restrict__ out) {
  __shared__ _Float16 smem[2 * BUFH];
  const int tid = threadIdx.x;
  const int lane = tid & 31;
  const int wave = tid >> 5;
  const int hi = lane >> 4;
  const int lo = lane & 15;
  const int n0 = blockIdx.x * 128 + wave * 16;
  const int m_base = blockIdx.y * 128;

  v8f acc[8];
#pragma unroll
  for (int i = 0; i < 8; ++i) acc[i] = (v8f)0.f;

  gemm_core<CB_>(A, Bm, smem, m_base, n0, tid, acc);

  const int nn = n0 + lo;
  const int b = nn / T_;
  const int t = nn % T_;
#pragma unroll
  for (int mt = 0; mt < 8; ++mt) {
    const int ob = m_base + mt * 16 + hi * 8;
#pragma unroll
    for (int j = 0; j < 8; ++j) {
      const int o = ob + j;
      const size_t idx = ((size_t)b * C_ + o) * T_ + t;
      out[idx] = acc[mt][j] + bias[o] + xres[idx];
    }
  }
}

// ---------------------------------------------------------------------------
// Host launcher
// ---------------------------------------------------------------------------
extern "C" void kernel_launch(void* const* d_in, const int* in_sizes, int n_in,
                              void* d_out, int out_size, void* d_ws, size_t ws_size,
                              hipStream_t stream) {
  const float* x      = (const float*)d_in[0];
  const float* gamma  = (const float*)d_in[1];
  const float* beta   = (const float*)d_in[2];
  const float* w_down = (const float*)d_in[3];
  const float* b_down = (const float*)d_in[4];
  const float* w_up   = (const float*)d_in[5];
  const float* b_up   = (const float*)d_in[6];
  float* out = (float*)d_out;

  char* ws = (char*)d_ws;
  size_t off = 0;
  float* mr = (float*)(ws + off);            off += 4096;                       // [512][2]
  _Float16* w16d = (_Float16*)(ws + off);    off += (size_t)CB_ * C_ * 2;       // 512 KB
  _Float16* w16u = (_Float16*)(ws + off);    off += (size_t)C_ * CB_ * 2;       // 512 KB
  _Float16* hn   = (_Float16*)(ws + off);    off += (size_t)NTOT * C_ * 2;      // 64 MB
  _Float16* d1   = (_Float16*)(ws + off);    off += (size_t)NTOT * CB_ * 2;     // 16 MB

  // 1) GroupNorm stats
  gn_stats_kernel<<<B_ * G_, 256, 0, stream>>>(x, mr);

  // 3) weight casts (independent of stats)
  cvt_f16_kernel<<<(CB_ * C_ + 255) / 256, 256, 0, stream>>>(w_down, w16d, CB_ * C_);
  cvt_f16_kernel<<<(C_ * CB_ + 255) / 256, 256, 0, stream>>>(w_up, w16u, C_ * CB_);

  // 2) normalize + transpose to [B*T, C] f16
  gn_apply_t_kernel<<<dim3(T_ / 32, C_ / 32, B_), 256, 0, stream>>>(
      x, gamma, beta, mr, hn);

  // 4) down-projection + mish  ([32768 x 1024] x [1024 x 256])
  gemm_down_mish_kernel<<<dim3(NTOT / 128, CB_ / 128), 256, 0, stream>>>(
      w16d, hn, b_down, d1);

  // 5) up-projection + bias + residual ([32768 x 256] x [256 x 1024])
  gemm_up_res_kernel<<<dim3(NTOT / 128, C_ / 128), 256, 0, stream>>>(
      w16u, d1, b_up, x, out);
}
